// DistanceCorrelationLoss_70961449664957
// MI455X (gfx1250) — matrix operand order repaired and checked by reading, not compiled, not run on hardware
//
#include <hip/hip_runtime.h>
#include <hip/hip_bf16.h>
#include <math.h>

typedef __bf16 bf16;
typedef __attribute__((ext_vector_type(16))) __bf16 v16bf;
typedef __attribute__((ext_vector_type(8)))  __bf16 v8bf;
typedef __attribute__((ext_vector_type(8)))  float  v8f;

#define BATCH 512
#define NFEAT 128
#define NTRIP 1024
#define A_PER_BLOCK 4   // 128 blocks * 4 = 512 'a' rows

// ---------------------------------------------------------------------------
// Kernel 1: per-feature row means of D and grand mean.
//   m[b*128 + i] = (1/B) sum_a |x_i(a) - x_i(b)|   (stored b-major for
//   coalesced reads in the gram kernel), g[i] = mean_b m_i(b).
// One workgroup per feature; x_i cached in LDS.
// ---------------------------------------------------------------------------
__global__ __launch_bounds__(256)
void dcor_mean_kernel(const float* __restrict__ emb,
                      float* __restrict__ m, float* __restrict__ g) {
    __shared__ float xs[BATCH];
    __shared__ float red[256];
    const int i = blockIdx.x;
    const int t = threadIdx.x;
    for (int b = t; b < BATCH; b += 256) xs[b] = emb[b * NFEAT + i];
    __syncthreads();
    float local = 0.f;
    for (int b = t; b < BATCH; b += 256) {
        const float xb = xs[b];
        float s = 0.f;
        #pragma unroll 4
        for (int a = 0; a < BATCH; ++a) s += fabsf(xs[a] - xb);
        const float mv = s * (1.0f / (float)BATCH);
        m[b * NFEAT + i] = mv;
        local += mv;
    }
    red[t] = local;
    __syncthreads();
    for (int s2 = 128; s2 > 0; s2 >>= 1) {
        if (t < s2) red[t] += red[t + s2];
        __syncthreads();
    }
    if (t == 0) g[i] = red[0] * (1.0f / (float)BATCH);
}

// ---------------------------------------------------------------------------
// Kernel 2: S += sum_k A[i][k] * A[j][k], k = (a,b) flattened, built on the
// fly in bf16 into LDS, consumed by v_wmma_f32_16x16x32_bf16.
// Block = 256 threads = 8 waves. Wave w owns output rows [16w, 16w+16).
// Each block handles A_PER_BLOCK 'a' rows x 512 b's = 2048 k slices.
// ---------------------------------------------------------------------------
__global__ __launch_bounds__(256)
void dcor_gram_kernel(const float* __restrict__ emb,
                      const float* __restrict__ m,
                      const float* __restrict__ g,
                      float* __restrict__ S) {
    // 128 feature rows x 32 k, padded to 40 bf16 (80 B) per row.
    __shared__ __align__(16) bf16 Abuf[NFEAT][40];

    const int t    = threadIdx.x;
    const int w    = t >> 5;        // wave id 0..7
    const int lane = t & 31;
    const int ifeat = t & 127;      // fill-phase feature (coalesced over emb)
    const int bhalf = (t >> 7) * 16;

    v8f c[8] = {};                  // 8 accum tiles per wave (rows 16w..)

    // Per-lane fragment addressing (constant across loop):
    const int arow = 16 * w + (lane & 15);     // A frag row
    const int akb  = (lane >> 4) * 8;          // A frag K base
    const int bkb  = (lane >> 4) * 8;          // B frag K base
    const float gi = g[ifeat];

    const int a0 = blockIdx.x * A_PER_BLOCK;
    for (int aa = 0; aa < A_PER_BLOCK; ++aa) {
        const int a = a0 + aa;
        const float xa = emb[a * NFEAT + ifeat];
        const float ma = m[a * NFEAT + ifeat];
        for (int bc = 0; bc < BATCH / 32; ++bc) {
            // ---- fill phase: 128x32 bf16 A tile, 16 elements per thread ----
            const int bbase = bc * 32 + bhalf;
            #pragma unroll
            for (int bb = 0; bb < 16; ++bb) {
                const int b = bbase + bb;
                const float xb = emb[b * NFEAT + ifeat];
                const float mb = m[b * NFEAT + ifeat];
                const float Av = fabsf(xa - xb) - ma - mb + gi;
                Abuf[ifeat][bhalf + bb] = (bf16)Av;
            }
            __syncthreads();
            // ---- WMMA phase (EXEC all ones, no divergence) ----
            union { v16bf v; v8bf h[2]; } af;
            af.h[0] = *(const v8bf*)&Abuf[arow][akb];        // K akb..akb+7
            af.h[1] = *(const v8bf*)&Abuf[arow][akb + 16];   // K akb+16..akb+23
            #pragma unroll
            for (int nj = 0; nj < 8; ++nj) {
                const int brow = 16 * nj + (lane & 15);
                union { v16bf v; v8bf h[2]; } bfv;
                bfv.h[0] = *(const v8bf*)&Abuf[brow][bkb];       // K bkb..bkb+7
                bfv.h[1] = *(const v8bf*)&Abuf[brow][bkb + 16];  // K bkb+16..bkb+23
                c[nj] = __builtin_amdgcn_wmma_f32_16x16x32_bf16(
                    false, af.v, false, bfv.v, (short)0, c[nj], false, false);
            }
            __syncthreads();
        }
    }
    // ---- epilogue: C/D layout -> global atomic accumulate ----
    const int rowbase = 16 * w + 8 * (lane >> 4);
    const int colbase = lane & 15;
    #pragma unroll
    for (int nj = 0; nj < 8; ++nj) {
        #pragma unroll
        for (int r = 0; r < 8; ++r) {
            atomicAdd(&S[(rowbase + r) * NFEAT + 16 * nj + colbase], c[nj][r]);
        }
    }
}

// ---------------------------------------------------------------------------
// Kernel 3: triplet margin loss. One wave32 per triplet, shfl reduction.
// ---------------------------------------------------------------------------
__global__ __launch_bounds__(256)
void triplet_kernel(const float* __restrict__ emb,
                    const long long* __restrict__ trip,
                    float* __restrict__ acc) {
    const int w    = threadIdx.x >> 5;
    const int lane = threadIdx.x & 31;
    const int tr   = blockIdx.x * 8 + w;
    const long long ia = trip[tr * 3 + 0];
    const long long ip = trip[tr * 3 + 1];
    const long long in_ = trip[tr * 3 + 2];
    float ap = 0.f, an = 0.f;
    #pragma unroll
    for (int f = lane; f < NFEAT; f += 32) {
        const float p  = emb[ia * NFEAT + f];
        const float dp = p - emb[ip * NFEAT + f];
        const float dn = p - emb[in_ * NFEAT + f];
        ap += dp * dp;
        an += dn * dn;
    }
    #pragma unroll
    for (int msk = 16; msk > 0; msk >>= 1) {
        ap += __shfl_xor(ap, msk, 32);
        an += __shfl_xor(an, msk, 32);
    }
    if (lane == 0) {
        const float l = ap - an + 1.0f;   // MARGIN = 1.0
        if (l > 0.f) atomicAdd(acc, l * (1.0f / (float)NTRIP));
    }
}

// ---------------------------------------------------------------------------
// Kernel 4: corr = sqrt(S/B^2) / (diag^1/4 * diag^1/4); mean over strict
// upper triangle; add triplet loss; write scalar.
// ---------------------------------------------------------------------------
__global__ __launch_bounds__(256)
void dcor_final_kernel(const float* __restrict__ S,
                       const float* __restrict__ trip_acc,
                       float* __restrict__ out) {
    __shared__ float q[NFEAT];
    __shared__ float red[256];
    const int t = threadIdx.x;
    const float invB2 = 1.0f / ((float)BATCH * (float)BATCH);
    if (t < NFEAT) q[t] = sqrtf(sqrtf(S[t * NFEAT + t] * invB2));
    __syncthreads();
    float acc = 0.f;
    for (int idx = t; idx < NFEAT * NFEAT; idx += 256) {
        const int i = idx >> 7;
        const int j = idx & 127;
        if (j > i) acc += sqrtf(S[idx] * invB2) / (q[i] * q[j]);
    }
    red[t] = acc;
    __syncthreads();
    for (int s = 128; s > 0; s >>= 1) {
        if (t < s) red[t] += red[t + s];
        __syncthreads();
    }
    if (t == 0)
        out[0] = red[0] * (2.0f / ((float)NFEAT * (NFEAT - 1))) + trip_acc[0];
}

// ---------------------------------------------------------------------------
extern "C" void kernel_launch(void* const* d_in, const int* in_sizes, int n_in,
                              void* d_out, int out_size, void* d_ws, size_t ws_size,
                              hipStream_t stream) {
    const float*     emb  = (const float*)d_in[0];
    const long long* trip = (const long long*)d_in[1];

    float* ws       = (float*)d_ws;
    float* m_ws     = ws;                              // 512*128 = 65536 floats
    float* g_ws     = ws + 65536;                      // 128 floats
    float* S_ws     = ws + 65536 + 128;                // 128*128 = 16384 floats
    float* trip_acc = ws + 65536 + 128 + 16384;        // 1 float (adjacent to S)

    // Zero the accumulators every launch (ws is not re-poisoned between replays).
    hipMemsetAsync(S_ws, 0, (16384 + 1) * sizeof(float), stream);

    dcor_mean_kernel<<<NFEAT, 256, 0, stream>>>(emb, m_ws, g_ws);
    dcor_gram_kernel<<<BATCH / A_PER_BLOCK, 256, 0, stream>>>(emb, m_ws, g_ws, S_ws);
    triplet_kernel<<<NTRIP / 8, 256, 0, stream>>>(emb, trip, trip_acc);
    dcor_final_kernel<<<1, 256, 0, stream>>>(S_ws, trip_acc, (float*)d_out);
}
